// Model_81432579932841
// MI455X (gfx1250) — matrix operand order repaired and checked by reading, not compiled, not run on hardware
//
#include <hip/hip_runtime.h>
#include <hip/hip_bf16.h>

typedef __attribute__((ext_vector_type(16))) __bf16 v16bf;
typedef __attribute__((ext_vector_type(8)))  __bf16 v8bf;
typedef __attribute__((ext_vector_type(8)))  float  v8f;

static __device__ __forceinline__ float log2pi() { return 1.8378770664093453f; }

#define B_  32
#define T_  1024
#define S_  256
#define N_  80
#define KP_ 96   // N padded to multiple of 32 for bf16 WMMA K-steps

// ---------------------------------------------------------------------------
// fp32 -> bf16 convert with K-padding (zeros), row-major [rows, KP]
// ---------------------------------------------------------------------------
__global__ void convert_pad_bf16(const float* __restrict__ in, __bf16* __restrict__ out,
                                 int rows, int K, int Kp) {
    int i = blockIdx.x * blockDim.x + threadIdx.x;
    if (i >= rows * Kp) return;
    int r = i / Kp, k = i - r * Kp;
    float v = (k < K) ? in[(size_t)r * K + k] : 0.0f;
    out[i] = (__bf16)v;
}

// ---------------------------------------------------------------------------
// exact fp32 row squared-norms
// ---------------------------------------------------------------------------
__global__ void row_sqnorm(const float* __restrict__ in, float* __restrict__ out,
                           int rows, int K) {
    int r = blockIdx.x * blockDim.x + threadIdx.x;
    if (r >= rows) return;
    const float* p = in + (size_t)r * K;
    float s = 0.f;
    for (int k = 0; k < K; ++k) s = fmaf(p[k], p[k], s);
    out[r] = s;
}

// ---------------------------------------------------------------------------
// ll[b,t,s] = -0.5*(log2pi + |x|^2 - 2 x.m + |m|^2) * attnmask
// Cross term via v_wmma_f32_16x16x32_bf16. One block = one 16-row t-tile
// across all S=256 columns; 4 waves, each wave owns 4 s-tiles (64 cols).
// ---------------------------------------------------------------------------
__global__ void ll_wmma_kernel(const __bf16* __restrict__ latentbf,
                               const __bf16* __restrict__ meanbf,
                               const float*  __restrict__ xnorm,
                               const float*  __restrict__ mnorm,
                               const int*    __restrict__ textlen,
                               const int*    __restrict__ mellen,
                               float*        __restrict__ ll) {
    const int tilesPerB = T_ / 16;
    int b  = blockIdx.x / tilesPerB;
    int t0 = (blockIdx.x % tilesPerB) * 16;
    int w    = threadIdx.x >> 5;     // wave 0..3
    int lane = threadIdx.x & 31;
    int lr   = lane & 15;
    bool hi  = lane >= 16;
    int text = textlen[b];
    int mel  = mellen[b];

    // A fragment base: row M = lr; lanes<16 take K 0..15, hi lanes 16..31
    const __bf16* abase = latentbf + ((size_t)(b * T_ + t0 + lr)) * KP_ + (hi ? 16 : 0);

    v8f acc[4] = {};
    #pragma unroll
    for (int kk = 0; kk < 3; ++kk) {          // K = 96 = 3 x 32
        v8bf alo = *(const v8bf*)(abase + kk * 32);
        v8bf ahi = *(const v8bf*)(abase + kk * 32 + 8);
        v16bf a;
        #pragma unroll
        for (int i = 0; i < 8; ++i) { a[i] = alo[i]; a[i + 8] = ahi[i]; }
        #pragma unroll
        for (int st = 0; st < 4; ++st) {
            int s0 = w * 64 + st * 16;
            // B fragment: column N = lr is mean row (s0+lr); lanes<16 K0..15, hi K16..31
            const __bf16* bptr = meanbf + ((size_t)(b * S_ + s0 + lr)) * KP_
                                        + kk * 32 + (hi ? 16 : 0);
            v16bf bm = *(const v16bf*)bptr;
            acc[st] = __builtin_amdgcn_wmma_f32_16x16x32_bf16(
                false, a, false, bm, (short)0, acc[st], false, false);
        }
    }

    // Epilogue: C layout — VGPR v: row = v + (hi?8:0), col = lr
    #pragma unroll
    for (int st = 0; st < 4; ++st) {
        int s = w * 64 + st * 16 + lr;
        float mn = mnorm[b * S_ + s];
        bool sok = (s < text);
        #pragma unroll
        for (int v = 0; v < 8; ++v) {
            int t = t0 + v + (hi ? 8 : 0);
            float xn   = xnorm[b * T_ + t];
            float dist = xn - 2.0f * acc[st][v] + mn;
            float val  = -0.5f * (log2pi() + dist);
            if (!(sok && (t < mel))) val = 0.0f;
            ll[((size_t)b * T_ + t) * S_ + s] = val;
        }
    }
}

// ---------------------------------------------------------------------------
// Viterbi forward: one block per batch, thread s owns column s.
// prob[s] <- max(prob[s], prob[s-1]) + ll[j,s]  (NEG outside s<=j)
// Double-buffered LDS prob -> one barrier per step. Rows j >= mellen have
// direction forced to 1 by the mask, so the recurrence stops at j = mellen
// and those rows are filled with 0xFFFFFFFF in parallel.
// ---------------------------------------------------------------------------
__global__ void mas_forward(const float* __restrict__ ll, unsigned* __restrict__ dirp,
                            const int* __restrict__ textlen, const int* __restrict__ mellen) {
    int b = blockIdx.x;
    int s = threadIdx.x;
    __shared__ float probsh[2][S_];
    const float NEGF = -3.402823466e38f;
    int text = textlen[b], mel = mellen[b];
    const float* llb  = ll   + (size_t)b * T_ * S_;
    unsigned*    dirb = dirp + (size_t)b * T_ * 8;

    // masked tail rows: direction == 1 everywhere
    for (int i = (size_t)mel * 8 + s; i < T_ * 8; i += S_) dirb[i] = 0xFFFFFFFFu;

    float prob = 0.0f;
    probsh[0][s] = 0.0f;
    __syncthreads();

    float llj = llb[s];                        // row j=0 prefetched
    for (int j = 0; j < mel; ++j) {
        int cur = j & 1, nxt = cur ^ 1;
        float llnext = (j + 1 < mel) ? llb[(size_t)(j + 1) * S_ + s] : 0.0f; // hide latency
        float prev = (s == 0) ? NEGF : probsh[cur][s - 1];
        bool  stay = prob >= prev;
        float pm   = stay ? prob : prev;
        float pn   = (s <= j) ? (pm + llj) : NEGF;
        bool  bit  = (s < text) ? stay : true;            // j < mel here
        unsigned long long m = __ballot(bit);             // wave32: low 32 bits
        if ((s & 31) == 0) dirb[(size_t)j * 8 + (s >> 5)] = (unsigned)m;
        probsh[nxt][s] = pn;
        prob = pn;
        llj = llnext;
        __syncthreads();
    }
}

// ---------------------------------------------------------------------------
// Backtrack (LDS-resident pointer chase, only mellen steps) + attn scatter +
// residual gather. One block per batch, 256 threads. dir bits staged to LDS.
// ---------------------------------------------------------------------------
__global__ void mas_backtrack_out(const unsigned* __restrict__ dirp,
                                  const float* __restrict__ latent,
                                  const float* __restrict__ mean,
                                  const int* __restrict__ textlen,
                                  const int* __restrict__ mellen,
                                  float* __restrict__ residual,
                                  float* __restrict__ attn) {
    int b = blockIdx.x, tid = threadIdx.x;
    __shared__ unsigned dirsh[T_ * 8];
    __shared__ short    idxsh[T_];

    int text = textlen[b], mel = mellen[b];

    const unsigned* dirb = dirp + (size_t)b * T_ * 8;
    for (int i = tid; i < mel * 8; i += 256) dirsh[i] = dirb[i];
    // j >= mel: direction all-ones -> idx pinned at text-1
    for (int j = mel + tid; j < T_; j += 256) idxsh[j] = (short)(text - 1);
    __syncthreads();

    if (tid == 0) {
        int idx = text - 1;
        for (int j = mel - 1; j >= 0; --j) {
            idxsh[j] = (short)idx;
            unsigned wv = dirsh[j * 8 + (idx >> 5)];
            int bit = (wv >> (idx & 31)) & 1;
            idx += bit - 1;               // stay (1) or move to s-1 (0)
            if (idx < 0) idx = 0;
        }
    }
    __syncthreads();

    // attn: zero slice, then scatter one-hot (masked)
    float* attnb = attn + (size_t)b * T_ * S_;
    for (int i = tid; i < T_ * S_; i += 256) attnb[i] = 0.0f;
    __syncthreads();
    for (int t = tid; t < T_; t += 256)
        if (t < mel) attnb[(size_t)t * S_ + (int)idxsh[t]] = 1.0f;

    // residual = latent - (t<mel ? mean[b, idx[t], :] : 0)
    const float* latb  = latent + (size_t)b * T_ * N_;
    const float* meanb = mean   + (size_t)b * S_ * N_;
    float*       resb  = residual + (size_t)b * T_ * N_;
    for (int i = tid; i < T_ * N_; i += 256) {
        int t = i / N_, n = i - t * N_;
        float al = (t < mel) ? meanb[(size_t)((int)idxsh[t]) * N_ + n] : 0.0f;
        resb[i] = latb[i] - al;
    }
}

// ---------------------------------------------------------------------------
extern "C" void kernel_launch(void* const* d_in, const int* in_sizes, int n_in,
                              void* d_out, int out_size, void* d_ws, size_t ws_size,
                              hipStream_t stream) {
    const float* latent  = (const float*)d_in[0];   // [B,T,N]
    const float* mean    = (const float*)d_in[1];   // [B,S,N]
    const int*   textlen = (const int*)d_in[2];     // [B]
    const int*   mellen  = (const int*)d_in[3];     // [B]

    float* residual = (float*)d_out;                        // [B,T,N]
    float* attn     = (float*)d_out + (size_t)B_ * T_ * N_; // [B,T,S]

    char* ws = (char*)d_ws;
    float*    llbuf    = (float*)   (ws);                                   // 32 MB
    unsigned* dirp     = (unsigned*)(ws + (size_t)B_ * T_ * S_ * 4);        // 1 MB
    __bf16*   latentbf = (__bf16*)  (ws + (size_t)B_ * T_ * S_ * 4
                                        + (size_t)B_ * T_ * 8 * 4);
    __bf16*   meanbf   = (__bf16*)  ((char*)latentbf + (size_t)B_ * T_ * KP_ * 2);
    float*    xnorm    = (float*)   ((char*)meanbf   + (size_t)B_ * S_ * KP_ * 2);
    float*    mnorm    = (float*)   ((char*)xnorm    + (size_t)B_ * T_ * 4);

    // 1) bf16 convert + pad
    {
        int n1 = B_ * T_ * KP_;
        convert_pad_bf16<<<(n1 + 255) / 256, 256, 0, stream>>>(latent, latentbf, B_ * T_, N_, KP_);
        int n2 = B_ * S_ * KP_;
        convert_pad_bf16<<<(n2 + 255) / 256, 256, 0, stream>>>(mean, meanbf, B_ * S_, N_, KP_);
    }
    // 2) exact fp32 row norms
    row_sqnorm<<<(B_ * T_ + 255) / 256, 256, 0, stream>>>(latent, xnorm, B_ * T_, N_);
    row_sqnorm<<<(B_ * S_ + 255) / 256, 256, 0, stream>>>(mean,   mnorm, B_ * S_, N_);
    // 3) WMMA log-likelihood
    ll_wmma_kernel<<<B_ * (T_ / 16), 128, 0, stream>>>(latentbf, meanbf, xnorm, mnorm,
                                                       textlen, mellen, llbuf);
    // 4) Viterbi forward (per-batch recurrence, mellen steps)
    mas_forward<<<B_, S_, 0, stream>>>(llbuf, dirp, textlen, mellen);
    // 5) Backtrack + outputs
    mas_backtrack_out<<<B_, 256, 0, stream>>>(dirp, latent, mean, textlen, mellen,
                                              residual, attn);
}
